// MultiHeadSelfAttention_85624468013578
// MI455X (gfx1250) — matrix operand order repaired and checked by reading, not compiled, not run on hardware
//
#include <hip/hip_runtime.h>
#include <hip/hip_bf16.h>

typedef __attribute__((ext_vector_type(16))) __bf16 v16bf;
typedef __attribute__((ext_vector_type(8)))  float  v8f;

namespace {
constexpr int kHeads = 16;
constexpr int kUnits = 1024;
constexpr int kHd    = 64;                 // head dim
constexpr int kSeq   = 2048;
constexpr int kBatch = 4;
constexpr int kRow   = 3 * kUnits;         // 3072 floats per token row (q|k|v)
constexpr int kWaves = 8;                  // waves per block (wave32)
constexpr int kBlkQ  = kWaves * 16;        // 128 query rows per block
constexpr int kQB    = kSeq / kBlkQ;       // 16 q-blocks per (b,h)
constexpr int kKB    = 64;                 // keys per inner step (2 WMMA K-chunks)
constexpr int kNKB   = kSeq / kKB;         // 32 steps
constexpr int kKStr  = 72;                 // padded LDS row stride (ushort), K tile rows
constexpr int kVStr  = 72;                 // padded LDS row stride, V^T rows (64 keys)
constexpr int kPStr  = 72;                 // padded LDS row stride, P scratch rows
// exp2-domain constants: softmax(x) == softmax(log2e * x)
constexpr float kScale2 = 0.125f * 1.4426950408889634f;  // (1/sqrt(64)) * log2e
constexpr float kNegBig = 1.0e9f * 1.4426950408889634f;  // 1e9 * log2e
}

union BF16x16 { v16bf v; uint4 q[2]; unsigned short s[16]; };
union F4 { float4 v; float f[4]; };

__device__ __forceinline__ unsigned short f2bf(float f) {
  __bf16 h = (__bf16)f;
  return __builtin_bit_cast(unsigned short, h);
}
__device__ __forceinline__ unsigned int pack2bf(float a, float b) {
  return (unsigned int)f2bf(a) | ((unsigned int)f2bf(b) << 16);
}
__device__ __forceinline__ float fast_exp2(float x) {
#if __has_builtin(__builtin_amdgcn_exp2f)
  return __builtin_amdgcn_exp2f(x);   // bare v_exp_f32
#else
  return exp2f(x);
#endif
}
// 32 contiguous bytes of LDS bf16 -> one WMMA B operand (16 bf16 per lane)
__device__ __forceinline__ v16bf lds_bop(const unsigned short* p) {
  BF16x16 t;
  const uint4* q = (const uint4*)p;
  t.q[0] = q[0];
  t.q[1] = q[1];
  return t.v;
}

__global__ __launch_bounds__(kWaves * 32) void
mhsa_flash_wmma(const float* __restrict__ in, const int* __restrict__ mask,
                float* __restrict__ out) {
  // Double-buffered K / V^T tiles (ping-pong, one barrier per step).
  __shared__ __align__(16) unsigned short kT[2][kKB * kKStr];
  __shared__ __align__(16) unsigned short vT[2][kHd * kVStr];
  __shared__ __align__(16) unsigned short pS[kWaves * 16 * kPStr];
  __shared__ float mF[kSeq];

  const int tid  = threadIdx.x;
  const int wave = tid >> 5;
  const int lane = tid & 31;
  const int hh   = lane >> 4;   // half-wave selector
  const int ln   = lane & 15;
  const int koff = hh * 16;     // A-layout K sub-offset

  const int qb = blockIdx.x % kQB;
  const int bh = blockIdx.x / kQB;
  const int h  = bh % kHeads;
  const int b  = bh / kHeads;

  // Stage this batch's mask row as floats (reused by all key steps).
  const int* mrow = mask + b * kSeq;
  for (int i = tid; i < kSeq; i += kWaves * 32) mF[i] = (float)mrow[i];

  // ---- Q (16 rows x 64 dims) straight into WMMA A-layout registers ----
  const int qrow0 = qb * kBlkQ + wave * 16;
  const float* qptr = in + ((size_t)(b * kSeq + qrow0 + ln)) * kRow + h * kHd;
  BF16x16 aQ[2];
#pragma unroll
  for (int c = 0; c < 2; ++c) {
    const float4 f0 = *(const float4*)(qptr + c * 32 + koff);
    const float4 f1 = *(const float4*)(qptr + c * 32 + koff + 4);
    const float4 g0 = *(const float4*)(qptr + c * 32 + 8 + koff);
    const float4 g1 = *(const float4*)(qptr + c * 32 + 8 + koff + 4);
    aQ[c].s[0]  = f2bf(f0.x); aQ[c].s[1]  = f2bf(f0.y);
    aQ[c].s[2]  = f2bf(f0.z); aQ[c].s[3]  = f2bf(f0.w);
    aQ[c].s[4]  = f2bf(f1.x); aQ[c].s[5]  = f2bf(f1.y);
    aQ[c].s[6]  = f2bf(f1.z); aQ[c].s[7]  = f2bf(f1.w);
    aQ[c].s[8]  = f2bf(g0.x); aQ[c].s[9]  = f2bf(g0.y);
    aQ[c].s[10] = f2bf(g0.z); aQ[c].s[11] = f2bf(g0.w);
    aQ[c].s[12] = f2bf(g1.x); aQ[c].s[13] = f2bf(g1.y);
    aQ[c].s[14] = f2bf(g1.z); aQ[c].s[15] = f2bf(g1.w);
  }

  // Ones B-operand: P x ones gives per-row sums replicated across all lanes.
  BF16x16 onesB;
#pragma unroll
  for (int i = 0; i < 16; ++i) onesB.s[i] = 0x3F80;  // bf16(1.0)

  // o[0..3] = 16x64 output accumulator, o[4] = running row-sum.
  v8f o[5] = {v8f{}, v8f{}, v8f{}, v8f{}, v8f{}};
  float rmax[8];
#pragma unroll
  for (int r = 0; r < 8; ++r) rmax[r] = -1e30f;

  const float* kbase = in + (size_t)b * kSeq * kRow + kUnits + h * kHd;
  const float* vbase = kbase + kUnits;
  unsigned short* ps = &pS[wave * 16 * kPStr];

  // Loader mappings (per thread):
  //  K: 4 float4 tiles, key = f4>>4, dm = (f4&15)*4        -> 2 packed b32 LDS stores each
  //  V: 2 (2-key x 4-dim) tiles, kp = tt>>4, dp4 = tt&15   -> 4 packed b32 LDS stores each
  F4 kreg[4], vra[2], vrb[2];

  auto ld_tile = [&](int kb) {
#pragma unroll
    for (int it = 0; it < 4; ++it) {
      const int f4  = it * (kWaves * 32) + tid;
      const int key = f4 >> 4;
      const int dm  = (f4 & 15) << 2;
      kreg[it].v = *(const float4*)(kbase + (size_t)(kb * kKB + key) * kRow + dm);
    }
#pragma unroll
    for (int it = 0; it < 2; ++it) {
      const int tt  = it * (kWaves * 32) + tid;
      const int kp  = tt >> 4;            // key pair 0..31
      const int d0  = (tt & 15) << 2;     // dim quad base
      vra[it].v = *(const float4*)(vbase + (size_t)(kb * kKB + 2 * kp) * kRow + d0);
      vrb[it].v = *(const float4*)(vbase + (size_t)(kb * kKB + 2 * kp + 1) * kRow + d0);
    }
  };
  auto st_tile = [&](int p) {
#pragma unroll
    for (int it = 0; it < 4; ++it) {
      const int f4  = it * (kWaves * 32) + tid;
      const int key = f4 >> 4;
      const int dm  = (f4 & 15) << 2;
      unsigned int* kd = (unsigned int*)&kT[p][key * kKStr + dm];
      kd[0] = pack2bf(kreg[it].f[0], kreg[it].f[1]);
      kd[1] = pack2bf(kreg[it].f[2], kreg[it].f[3]);
    }
#pragma unroll
    for (int it = 0; it < 2; ++it) {
      const int tt = it * (kWaves * 32) + tid;
      const int kp = tt >> 4;
      const int d0 = (tt & 15) << 2;
#pragma unroll
      for (int j = 0; j < 4; ++j) {   // V^T: two keys packed per dword
        *(unsigned int*)&vT[p][(d0 + j) * kVStr + 2 * kp] =
            pack2bf(vra[it].f[j], vrb[it].f[j]);
      }
    }
  };

  ld_tile(0);  // prime the pipeline

  for (int kb = 0; kb < kNKB; ++kb) {
    const int p = kb & 1;
    st_tile(p);          // staged registers -> LDS buffer p
    __syncthreads();     // single barrier per step (two-buffer hand-off)
    if (kb + 1 < kNKB) ld_tile(kb + 1);  // prefetch next tile under compute

    // ---- S = Q * K^T : four 16x16 f32 tiles (key groups of 16) ----
    v8f s[4] = {v8f{}, v8f{}, v8f{}, v8f{}};
#pragma unroll
    for (int c = 0; c < 2; ++c) {
#pragma unroll
      for (int g = 0; g < 4; ++g) {
        v16bf bk = lds_bop(&kT[p][(g * 16 + ln) * kKStr + c * 32 + hh * 16]);
        s[g] = __builtin_amdgcn_wmma_f32_16x16x32_bf16(false, aQ[c].v, false, bk,
                                                       (short)0, s[g], false, false);
      }
    }

    // Per-key-column mask coefficients: w = s*am + bm  (exp2 domain).
    float am[4], bm[4];
#pragma unroll
    for (int g = 0; g < 4; ++g) {
      const float m = mF[kb * kKB + g * 16 + ln];
      am[g] = m * kScale2;
      bm[g] = (m - 1.0f) * kNegBig;
    }

    // ---- online softmax: one max-butterfly per row per 64 keys ----
#pragma unroll
    for (int r = 0; r < 8; ++r) {
      const float w0 = fmaf(s[0][r], am[0], bm[0]);
      const float w1 = fmaf(s[1][r], am[1], bm[1]);
      const float w2 = fmaf(s[2][r], am[2], bm[2]);
      const float w3 = fmaf(s[3][r], am[3], bm[3]);
      float mx = fmaxf(fmaxf(w0, w1), fmaxf(w2, w3));
      mx = fmaxf(mx, __shfl_xor(mx, 1));
      mx = fmaxf(mx, __shfl_xor(mx, 2));
      mx = fmaxf(mx, __shfl_xor(mx, 4));
      mx = fmaxf(mx, __shfl_xor(mx, 8));
      const float nm   = fmaxf(rmax[r], mx);
      const float corr = fast_exp2(rmax[r] - nm);
      rmax[r] = nm;
      const unsigned int pp01 = pack2bf(fast_exp2(w0 - nm), fast_exp2(w1 - nm));
      const unsigned int pp23 = pack2bf(fast_exp2(w2 - nm), fast_exp2(w3 - nm));
      o[0][r] *= corr; o[1][r] *= corr; o[2][r] *= corr;
      o[3][r] *= corr; o[4][r] *= corr;
      const int row = r + 8 * hh;              // C-layout -> scratch (row-major)
      ps[row * kPStr + ln]      = (unsigned short)(pp01 & 0xffffu);
      ps[row * kPStr + 16 + ln] = (unsigned short)(pp01 >> 16);   // d16_hi store
      ps[row * kPStr + 32 + ln] = (unsigned short)(pp23 & 0xffffu);
      ps[row * kPStr + 48 + ln] = (unsigned short)(pp23 >> 16);
    }
    // Same-wave LDS ordering: drain DS before cross-lane readback.
    asm volatile("s_wait_dscnt 0" ::: "memory");

    // ---- O += P * V (two 32-key chunks), plus sum column via ones ----
#pragma unroll
    for (int kc = 0; kc < 2; ++kc) {
      BF16x16 aP;
      aP.q[0] = *(const uint4*)&ps[ln * kPStr + kc * 32 + koff];
      aP.q[1] = *(const uint4*)&ps[ln * kPStr + kc * 32 + 8 + koff];
#pragma unroll
      for (int oc = 0; oc < 4; ++oc) {
        v16bf bv = lds_bop(&vT[p][(oc * 16 + ln) * kVStr + kc * 32 + hh * 16]);
        o[oc] = __builtin_amdgcn_wmma_f32_16x16x32_bf16(false, aP.v, false, bv,
                                                        (short)0, o[oc], false, false);
      }
      o[4] = __builtin_amdgcn_wmma_f32_16x16x32_bf16(false, aP.v, false, onesB.v,
                                                     (short)0, o[4], false, false);
    }
  }

  // ---- epilogue: normalize by row sum (replicated in o[4]) and store ----
#pragma unroll
  for (int r = 0; r < 8; ++r) {
    const float inv = 1.0f / o[4][r];
    float* orow = out + ((size_t)(b * kSeq + qrow0 + r + 8 * hh)) * kUnits +
                  h * kHd + ln;
    orow[0]  = o[0][r] * inv;
    orow[16] = o[1][r] * inv;
    orow[32] = o[2][r] * inv;
    orow[48] = o[3][r] * inv;
  }
}

extern "C" void kernel_launch(void* const* d_in, const int* in_sizes, int n_in,
                              void* d_out, int out_size, void* d_ws, size_t ws_size,
                              hipStream_t stream) {
  (void)in_sizes; (void)n_in; (void)out_size; (void)d_ws; (void)ws_size;
  const float* in = (const float*)d_in[0];
  const int* mask = (const int*)d_in[1];
  float* out = (float*)d_out;
  dim3 grid(kBatch * kHeads * kQB);   // 4*16*16 = 1024 blocks
  dim3 block(kWaves * 32);            // 256 threads = 8 waves (wave32)
  mhsa_flash_wmma<<<grid, block, 0, stream>>>(in, mask, out);
}
